// DFlashDraftModel_91328184582842
// MI455X (gfx1250) — matrix operand order repaired and checked by reading, not compiled, not run on hardware
//
#include <hip/hip_runtime.h>

// ---------------- problem constants ----------------
#define BATCH  2
#define QQ     32
#define CTXC   2048
#define LL     4
#define HH     2048
#define NHn    16
#define NKVn   8
#define HDd    128
#define INTERi 6144
#define KVk    (CTXC + QQ)     // 2080
#define GG     (NHn / NKVn)    // 2
#define EPSf   1e-6f
#define THETAf 1000000.0f
#define SCALEf 0.08838834764831845f   // 128^-0.5

// ---------------- WMMA types ----------------
typedef __attribute__((ext_vector_type(16))) __bf16 v16bf;
typedef __attribute__((ext_vector_type(2)))  __bf16 v2bf;
typedef __attribute__((ext_vector_type(2)))  float  v2f;
typedef __attribute__((ext_vector_type(8)))  float  v8f;

union FragAB { v16bf v; unsigned int u[8]; };

// float2 -> packed bf16x2 dword; lowers to a single v_cvt_pk_bf16_f32.
__device__ __forceinline__ unsigned int bf16pack(float lo, float hi) {
  v2f f = {lo, hi};
  v2bf b = __builtin_convertvector(f, v2bf);
  return __builtin_bit_cast(unsigned int, b);
}

// ---------------- generic batched/strided WMMA GEMM ----------------
// C[z](m,n) = alpha * sum_k A[z](m,k) * B[z](k,n)  (+ R[z](m,n))
// A(m,k) = A[m*lda + k]  row-major.  B(k,n) = B[k*ldbK + n*ldbN].
// PATHB=1: ldbN==1 (coalesced along n).  PATHB=2: ldbK==1 (contiguous along k).
// REQUIREMENTS (all call sites satisfy): K % 32 == 0, N % 4 == 0,
// lda/ldbK/ldbN multiples of 4, base pointers 16B aligned.
// batch: z = zo*batchInner + zi ; A/C use (zo,zi) strides, B uses zi/gDiv.
#define BM 64
#define BN 128
#define BK 64

struct GemmP {
  const float* A; const float* Bm; float* C; const float* R;
  int M, N, K;
  long long lda, ldbK, ldbN, ldc;
  float alpha;
  int batchInner, gDiv;
  long long sAo, sAi, sBo, sBi, sCo, sCi;
};

template <int PATHB>
__global__ __launch_bounds__(256) void wmma_gemm_kernel(GemmP p) {
  __shared__ unsigned short As[2][BM][BK + 2];   // bf16 bits, A-tile (m,k)
  __shared__ unsigned short Bs[2][BN][BK + 2];   // bf16 bits, B-tile stored (n,k)

  const int z  = blockIdx.z;
  const int zo = z / p.batchInner, zi = z % p.batchInner;
  const float* A  = p.A  + (long long)zo * p.sAo + (long long)zi * p.sAi;
  const float* Bg = p.Bm + (long long)zo * p.sBo + (long long)(zi / p.gDiv) * p.sBi;
  float*       C  = p.C  + (long long)zo * p.sCo + (long long)zi * p.sCi;
  const float* R  = p.R ? (p.R + (long long)zo * p.sCo + (long long)zi * p.sCi) : nullptr;

  const int m0 = blockIdx.y * BM;
  const int n0 = blockIdx.x * BN;
  const int tid  = threadIdx.x;
  const int lane = tid & 31;          // wave32
  const int wave = tid >> 5;          // 8 waves
  const int wm = (wave >> 2) * 32;    // 0 / 32
  const int wn = (wave & 3) * 32;    // 0 / 32 / 64 / 96

  // --- thread-fixed load geometry ---
  // A tile: thread -> (row, 16 consecutive k)
  const int arow = tid >> 2;                  // 0..63
  const int akk  = (tid & 3) * 16;            // 0,16,32,48
  const int agm  = m0 + arow;
  const float amask = (agm < p.M) ? 1.0f : 0.0f;
  const int   agmC  = (agm < p.M) ? agm : (p.M - 1);
  const float* aRow = A + (long long)agmC * p.lda;
  // path 2 geometry: thread -> (n, 32 consecutive k)
  const int p2n  = tid >> 1;
  const int p2kh = (tid & 1) * 32;

  float areg[16];
  float breg[32];

  auto loadT = [&](int k0) {
    // ---- A: 4x float4, clamped + masked (M and K guards, branch-free) ----
#pragma unroll
    for (int i = 0; i < 4; ++i) {
      const int kg = k0 + akk + 4 * i;
      const bool kin = (kg + 4 <= p.K);        // exact: K % 4 == 0
      const int kgC = kin ? kg : (p.K - 4);
      const float msk = kin ? amask : 0.0f;
      float4 v = *(const float4*)(aRow + kgC);
      areg[4*i+0] = v.x * msk; areg[4*i+1] = v.y * msk;
      areg[4*i+2] = v.z * msk; areg[4*i+3] = v.w * msk;
    }
    // ---- B ----
    if (PATHB == 1) {            // B(k,n)=Bg[k*ldbK+n]: 2 adjacent k-rows x 4 n
#pragma unroll
      for (int i = 0; i < 4; ++i) {
        const int g  = i * 256 + tid;
        const int k2 = (g >> 5) * 2;           // 0..62 even
        const int bn = n0 + (g & 31) * 4;
        const bool kin = (k0 + k2 + 2 <= p.K); // exact: K even
        const int kC = kin ? (k0 + k2) : (p.K - 2);
        const bool nin = (bn + 4 <= p.N);      // exact: N % 4 == 0
        const int bnC = nin ? bn : (p.N - 4);
        const float msk = (kin && nin) ? 1.0f : 0.0f;
        float4 v0 = *(const float4*)(Bg + (long long)kC * p.ldbK + bnC);
        float4 v1 = *(const float4*)(Bg + (long long)(kC + 1) * p.ldbK + bnC);
        breg[8*i+0] = v0.x * msk; breg[8*i+1] = v0.y * msk;
        breg[8*i+2] = v0.z * msk; breg[8*i+3] = v0.w * msk;
        breg[8*i+4] = v1.x * msk; breg[8*i+5] = v1.y * msk;
        breg[8*i+6] = v1.z * msk; breg[8*i+7] = v1.w * msk;
      }
    } else {                     // B(k,n)=Bg[k+n*ldbN]: contiguous along k
      const int gn = n0 + p2n;
      const bool nin = (gn < p.N);
      const int gnC = nin ? gn : (p.N - 1);
      const float* src = Bg + (long long)gnC * p.ldbN;
#pragma unroll
      for (int j = 0; j < 8; ++j) {
        const int kg = k0 + p2kh + 4 * j;
        const bool kin = (kg + 4 <= p.K);
        const int kgC = kin ? kg : (p.K - 4);
        const float msk = (kin && nin) ? 1.0f : 0.0f;
        float4 v = *(const float4*)(src + kgC);
        breg[4*j+0] = v.x * msk; breg[4*j+1] = v.y * msk;
        breg[4*j+2] = v.z * msk; breg[4*j+3] = v.w * msk;
      }
    }
  };

  auto storeT = [&](int bb) {
    unsigned int* ad = (unsigned int*)&As[bb][arow][akk];
#pragma unroll
    for (int i = 0; i < 8; ++i)
      ad[i] = bf16pack(areg[2*i], areg[2*i+1]);
    if (PATHB == 1) {
#pragma unroll
      for (int i = 0; i < 4; ++i) {
        const int g   = i * 256 + tid;
        const int k2  = (g >> 5) * 2;
        const int bn4 = (g & 31) * 4;
#pragma unroll
        for (int j = 0; j < 4; ++j)
          *(unsigned int*)&Bs[bb][bn4 + j][k2] =
              bf16pack(breg[8*i+j], breg[8*i+4+j]);   // (k, k+1) pair
      }
    } else {
      unsigned int* bd = (unsigned int*)&Bs[bb][p2n][p2kh];
#pragma unroll
      for (int q = 0; q < 16; ++q)
        bd[q] = bf16pack(breg[2*q], breg[2*q+1]);
    }
  };

  v8f zero = {0.f,0.f,0.f,0.f,0.f,0.f,0.f,0.f};
  v8f acc[2][2];
  acc[0][0] = zero; acc[0][1] = zero; acc[1][0] = zero; acc[1][1] = zero;

  const int nK = (p.K + BK - 1) / BK;

  // ---- software pipeline: global->regs for ks+1 overlaps WMMA on ks ----
  loadT(0);
  storeT(0);
  __syncthreads();
  int buf = 0;
  for (int ks = 0; ks < nK; ++ks) {
    const bool nxt = (ks + 1) < nK;
    if (nxt) loadT((ks + 1) * BK);

    // ---- two WMMA K-halves per LDS tile (ISA 7.12.2 wave32 bf16 layouts) ----
#pragma unroll
    for (int kh = 0; kh < 2; ++kh) {
      const int po = kh * 16;                  // dword-pair offset of K-half
      FragAB a[2], b[2];
#pragma unroll
      for (int t = 0; t < 2; ++t) {
        const int m = wm + t * 16 + (lane & 15);
        const unsigned int* rp = (const unsigned int*)&As[buf][m][0];
        const int kb = (lane >> 4) * 4;        // dword-pair base: 0 or 4
#pragma unroll
        for (int j = 0; j < 8; ++j)
          a[t].u[j] = rp[po + ((j >= 4) ? 8 : 0) + kb + (j & 3)];
      }
#pragma unroll
      for (int t = 0; t < 2; ++t) {
        const int n = wn + t * 16 + (lane & 15);
        const unsigned int* rp = (const unsigned int*)&Bs[buf][n][0];
        const int kb = (lane >> 4) * 4;        // dword-pair base: 0 or 4
#pragma unroll
        for (int j = 0; j < 8; ++j)
          b[t].u[j] = rp[po + ((j >= 4) ? 8 : 0) + kb + (j & 3)];
      }
#pragma unroll
      for (int ti = 0; ti < 2; ++ti)
#pragma unroll
        for (int tj = 0; tj < 2; ++tj)
          acc[ti][tj] = __builtin_amdgcn_wmma_f32_16x16x32_bf16(
              false, a[ti].v, false, b[tj].v, (short)0, acc[ti][tj], false, false);
    }

    if (nxt) storeT(buf ^ 1);
    __syncthreads();
    buf ^= 1;
  }

  // ---- epilogue: C layout -> (VGPR i: M=i / M=8+i; lanes 0-15/16-31) ----
#pragma unroll
  for (int ti = 0; ti < 2; ++ti)
#pragma unroll
    for (int tj = 0; tj < 2; ++tj)
#pragma unroll
      for (int i = 0; i < 8; ++i) {
        const int row = m0 + wm + ti * 16 + i + ((lane >= 16) ? 8 : 0);
        const int col = n0 + wn + tj * 16 + (lane & 15);
        if (row < p.M && col < p.N) {
          float v = p.alpha * acc[ti][tj][i];
          const long long off = (long long)row * p.ldc + col;
          if (R) v += R[off];
          C[off] = v;
        }
      }
}

// ---------------- RMSNorm (row-wise, optional second destination) ----------------
__global__ __launch_bounds__(256) void rmsnorm_kernel(
    const float* src, float* dst, float* dst2, const float* w,
    int rowLen, int rowsPerBatch,
    long long srcB, long long srcR, long long dstB, long long dstR,
    long long dst2B, long long dst2R) {
  const int row = blockIdx.x;
  const int b = row / rowsPerBatch, r = row % rowsPerBatch;
  const float* s = src + (long long)b * srcB + (long long)r * srcR;
  __shared__ float red[256];
  float ss = 0.f;
  for (int i = threadIdx.x; i < rowLen; i += 256) { float v = s[i]; ss += v * v; }
  red[threadIdx.x] = ss; __syncthreads();
  for (int o = 128; o > 0; o >>= 1) {
    if ((int)threadIdx.x < o) red[threadIdx.x] += red[threadIdx.x + o];
    __syncthreads();
  }
  const float scale = rsqrtf(red[0] / (float)rowLen + EPSf);
  float* d1 = dst + (long long)b * dstB + (long long)r * dstR;
  float* d2 = dst2 ? (dst2 + (long long)b * dst2B + (long long)r * dst2R) : nullptr;
  for (int i = threadIdx.x; i < rowLen; i += 256) {
    const float v = s[i] * scale * w[i];
    d1[i] = v;
    if (d2) d2[i] = v;
  }
}

// ---------------- fused QK RMSNorm + RoPE + head-major transpose ----------------
// src [B, S, NHx, HD] -> dst [B, NHx, S, HD]
__global__ __launch_bounds__(64) void qknorm_rope_kernel(
    const float* src, float* dst, const float* w, const int* pos_ids,
    int S, int NHx, int posOff) {
  const int row = blockIdx.x;
  const int hh = row % NHx;
  const int s  = (row / NHx) % S;
  const int b  = row / (NHx * S);
  const float* sr = src + (((long long)(b * S + s)) * NHx + hh) * HDd;
  float*       dr = dst + (((long long)(b * NHx + hh)) * S + s) * HDd;
  const int t = threadIdx.x;                 // 0..63
  const float x1 = sr[t], x2 = sr[t + 64];
  __shared__ float red[64];
  red[t] = x1 * x1 + x2 * x2; __syncthreads();
  for (int o = 32; o > 0; o >>= 1) {
    if (t < o) red[t] += red[t + o];
    __syncthreads();
  }
  const float scale = rsqrtf(red[0] / (float)HDd + EPSf);
  const float n1 = x1 * scale * w[t];
  const float n2 = x2 * scale * w[t + 64];
  const float pos = (float)pos_ids[(long long)b * KVk + posOff + s];
  const float inv = __expf(-((float)(2 * t) / (float)HDd) * __logf(THETAf));
  const float ang = pos * inv;
  float sn, cs;
  __sincosf(ang, &sn, &cs);
  dr[t]      = n1 * cs - n2 * sn;
  dr[t + 64] = n2 * cs + n1 * sn;
}

// ---------------- softmax over last dim (one block per row) ----------------
__global__ __launch_bounds__(256) void softmax_kernel(float* sc, int len) {
  float* row = sc + (long long)blockIdx.x * len;
  __shared__ float red[256];
  float m = -3.4e38f;
  for (int i = threadIdx.x; i < len; i += 256) m = fmaxf(m, row[i]);
  red[threadIdx.x] = m; __syncthreads();
  for (int o = 128; o > 0; o >>= 1) {
    if ((int)threadIdx.x < o) red[threadIdx.x] = fmaxf(red[threadIdx.x], red[threadIdx.x + o]);
    __syncthreads();
  }
  m = red[0]; __syncthreads();
  float sum = 0.f;
  for (int i = threadIdx.x; i < len; i += 256) {
    const float e = __expf(row[i] - m);
    row[i] = e; sum += e;
  }
  red[threadIdx.x] = sum; __syncthreads();
  for (int o = 128; o > 0; o >>= 1) {
    if ((int)threadIdx.x < o) red[threadIdx.x] += red[threadIdx.x + o];
    __syncthreads();
  }
  const float invs = 1.0f / red[0];
  for (int i = threadIdx.x; i < len; i += 256) row[i] *= invs;
}

// ---------------- [D0,D1,D2,D3] -> [D0,D2,D1,D3] ----------------
__global__ void transpose_0213(const float* src, float* dst,
                               int D1, int D2, int D3, long long total) {
  long long idx = (long long)blockIdx.x * 256 + threadIdx.x;
  if (idx >= total) return;
  const int d3 = (int)(idx % D3); long long t = idx / D3;
  const int d2 = (int)(t % D2); t /= D2;
  const int d1 = (int)(t % D1);
  const long long d0 = t / D1;
  dst[((d0 * D2 + d2) * (long long)D1 + d1) * D3 + d3] = src[idx];
}

// ---------------- SiLU(g) * u (in place into g) ----------------
__global__ void silu_mul_kernel(float* g, const float* u, long long n) {
  const long long i = (long long)blockIdx.x * 256 + threadIdx.x;
  if (i < n) {
    const float gv = g[i];
    g[i] = gv / (1.0f + __expf(-gv)) * u[i];
  }
}

// ---------------- host-side helpers ----------------
static inline int ceilDiv(long long a, long long b) { return (int)((a + b - 1) / b); }

static void launch_gemm(hipStream_t st,
                        const float* A, const float* B, float* C, const float* R,
                        int M, int N, int K,
                        long long lda, long long ldbK, long long ldbN, long long ldc,
                        float alpha,
                        int nbatch = 1, int batchInner = 1, int gDiv = 1,
                        long long sAo = 0, long long sAi = 0,
                        long long sBo = 0, long long sBi = 0,
                        long long sCo = 0, long long sCi = 0) {
  GemmP p;
  p.A = A; p.Bm = B; p.C = C; p.R = R;
  p.M = M; p.N = N; p.K = K;
  p.lda = lda; p.ldbK = ldbK; p.ldbN = ldbN; p.ldc = ldc;
  p.alpha = alpha;
  p.batchInner = batchInner; p.gDiv = gDiv;
  p.sAo = sAo; p.sAi = sAi; p.sBo = sBo; p.sBi = sBi; p.sCo = sCo; p.sCi = sCi;
  dim3 grid(ceilDiv(N, BN), ceilDiv(M, BM), nbatch);
  if (ldbN == 1)
    wmma_gemm_kernel<1><<<grid, dim3(256), 0, st>>>(p);
  else
    wmma_gemm_kernel<2><<<grid, dim3(256), 0, st>>>(p);
}

extern "C" void kernel_launch(void* const* d_in, const int* in_sizes, int n_in,
                              void* d_out, int out_size, void* d_ws, size_t ws_size,
                              hipStream_t stream) {
  const float* noise    = (const float*)d_in[0];   // [B,Q,H]
  const float* tgt_hid  = (const float*)d_in[1];   // [B,CTX,L*H]
  const int*   pos_ids  = (const int*)  d_in[2];   // [B,KV]
  const float* fc_w     = (const float*)d_in[3];   // [L*H, H]
  const float* hnorm_w  = (const float*)d_in[4];   // [H]
  const float* ln1_w    = (const float*)d_in[5];   // [L,H]
  const float* ln2_w    = (const float*)d_in[6];   // [L,H]
  const float* wq       = (const float*)d_in[7];   // [L,H,NH*HD]
  const float* wk       = (const float*)d_in[8];   // [L,H,NKV*HD]
  const float* wv       = (const float*)d_in[9];   // [L,H,NKV*HD]
  const float* wo       = (const float*)d_in[10];  // [L,NH*HD,H]
  const float* qn_w     = (const float*)d_in[11];  // [L,HD]
  const float* kn_w     = (const float*)d_in[12];  // [L,HD]
  const float* gate_w   = (const float*)d_in[13];  // [L,H,INTER]
  const float* up_w     = (const float*)d_in[14];  // [L,H,INTER]
  const float* down_w   = (const float*)d_in[15];  // [L,INTER,H]
  const float* fnorm_w  = (const float*)d_in[16];  // [H]
  float* out = (float*)d_out;                      // [B,Q,H] fp32

  // ---- workspace bump allocator ----
  char* wsp = (char*)d_ws;
  auto alloc = [&](long long nfloats) -> float* {
    float* p = (float*)wsp;
    wsp += ((nfloats * 4 + 255) / 256) * 256;
    return p;
  };
  float* thTmp  = alloc((long long)BATCH * CTXC * HH);        // fc output pre-norm
  float* kvin   = alloc((long long)BATCH * KVk * HH);         // [B,KV,H] concat(th, x)
  float* h      = alloc((long long)BATCH * QQ * HH);
  float* x      = alloc((long long)BATCH * QQ * HH);
  float* qpre   = alloc((long long)BATCH * QQ * NHn * HDd);
  float* qrot   = alloc((long long)BATCH * QQ * NHn * HDd);   // [B,NH,Q,HD]
  float* kpre   = alloc((long long)BATCH * KVk * NKVn * HDd);
  float* krot   = alloc((long long)BATCH * KVk * NKVn * HDd); // [B,NKV,KV,HD]
  float* vpre   = alloc((long long)BATCH * KVk * NKVn * HDd);
  float* vrot   = alloc((long long)BATCH * KVk * NKVn * HDd); // [B,NKV,KV,HD]
  float* scores = alloc((long long)BATCH * NHn * QQ * KVk);   // [B,NH,Q,KV]
  float* obuf   = alloc((long long)BATCH * NHn * QQ * HDd);   // [B,NH,Q,HD]
  float* o2     = alloc((long long)BATCH * QQ * NHn * HDd);   // [B,Q,NH*HD]
  float* gbuf   = alloc((long long)BATCH * QQ * INTERi);
  float* ubuf   = alloc((long long)BATCH * QQ * INTERi);

  // h = noise_embedding
  hipMemcpyAsync(h, noise, (size_t)BATCH * QQ * HH * sizeof(float),
                 hipMemcpyDeviceToDevice, stream);

  // th = rms_norm(target_hidden @ fc_w) -> kvin rows [0, CTX)
  launch_gemm(stream, tgt_hid, fc_w, thTmp, nullptr,
              BATCH * CTXC, HH, LL * HH,
              LL * HH, HH, 1, HH, 1.0f);
  rmsnorm_kernel<<<BATCH * CTXC, 256, 0, stream>>>(
      thTmp, kvin, nullptr, hnorm_w, HH, CTXC,
      (long long)CTXC * HH, HH, (long long)KVk * HH, HH, 0, 0);

  const long long kvElems = (long long)BATCH * KVk * NKVn * HDd;

  for (int i = 0; i < LL; ++i) {
    // x = rms_norm(h, ln1) ; also written into kvin rows [CTX, KV)
    rmsnorm_kernel<<<BATCH * QQ, 256, 0, stream>>>(
        h, x, kvin + (long long)CTXC * HH, ln1_w + (long long)i * HH, HH, QQ,
        (long long)QQ * HH, HH, (long long)QQ * HH, HH, (long long)KVk * HH, HH);

    // q = x @ wq[i]   (M=64, N=2048, K=2048)
    launch_gemm(stream, x, wq + (long long)i * HH * NHn * HDd, qpre, nullptr,
                BATCH * QQ, NHn * HDd, HH, HH, NHn * HDd, 1, NHn * HDd, 1.0f);
    qknorm_rope_kernel<<<BATCH * QQ * NHn, 64, 0, stream>>>(
        qpre, qrot, qn_w + (long long)i * HDd, pos_ids, QQ, NHn, CTXC);

    // k = kvin @ wk[i]  (M=4160, N=1024, K=2048)
    launch_gemm(stream, kvin, wk + (long long)i * HH * NKVn * HDd, kpre, nullptr,
                BATCH * KVk, NKVn * HDd, HH, HH, NKVn * HDd, 1, NKVn * HDd, 1.0f);
    qknorm_rope_kernel<<<BATCH * KVk * NKVn, 64, 0, stream>>>(
        kpre, krot, kn_w + (long long)i * HDd, pos_ids, KVk, NKVn, 0);

    // v = kvin @ wv[i]; transpose to [B,NKV,KV,HD]
    launch_gemm(stream, kvin, wv + (long long)i * HH * NKVn * HDd, vpre, nullptr,
                BATCH * KVk, NKVn * HDd, HH, HH, NKVn * HDd, 1, NKVn * HDd, 1.0f);
    transpose_0213<<<ceilDiv(kvElems, 256), 256, 0, stream>>>(
        vpre, vrot, KVk, NKVn, HDd, kvElems);

    // scores[b,nh] = SCALE * q[b,nh] @ k[b,nh/G]^T  (M=32, N=2080, K=128)
    launch_gemm(stream, qrot, krot, scores, nullptr,
                QQ, KVk, HDd,
                HDd, /*ldbK=*/1, /*ldbN=*/HDd, KVk, SCALEf,
                BATCH * NHn, NHn, GG,
                (long long)NHn * QQ * HDd, (long long)QQ * HDd,
                (long long)NKVn * KVk * HDd, (long long)KVk * HDd,
                (long long)NHn * QQ * KVk, (long long)QQ * KVk);
    softmax_kernel<<<BATCH * NHn * QQ, 256, 0, stream>>>(scores, KVk);

    // o[b,nh] = P @ V  (M=32, N=128, K=2080)
    launch_gemm(stream, scores, vrot, obuf, nullptr,
                QQ, HDd, KVk,
                KVk, /*ldbK=*/HDd, /*ldbN=*/1, HDd, 1.0f,
                BATCH * NHn, NHn, GG,
                (long long)NHn * QQ * KVk, (long long)QQ * KVk,
                (long long)NKVn * KVk * HDd, (long long)KVk * HDd,
                (long long)NHn * QQ * HDd, (long long)QQ * HDd);
    // [B,NH,Q,HD] -> [B,Q,NH*HD]
    const long long oElems = (long long)BATCH * NHn * QQ * HDd;
    transpose_0213<<<ceilDiv(oElems, 256), 256, 0, stream>>>(
        obuf, o2, NHn, QQ, HDd, oElems);

    // h = h + o2 @ wo[i]
    launch_gemm(stream, o2, wo + (long long)i * NHn * HDd * HH, h, h,
                BATCH * QQ, HH, NHn * HDd, NHn * HDd, HH, 1, HH, 1.0f);

    // x2 = rms_norm(h, ln2) (reuse x)
    rmsnorm_kernel<<<BATCH * QQ, 256, 0, stream>>>(
        h, x, nullptr, ln2_w + (long long)i * HH, HH, QQ,
        (long long)QQ * HH, HH, (long long)QQ * HH, HH, 0, 0);

    // MLP: h += (silu(x2@gate) * (x2@up)) @ down
    launch_gemm(stream, x, gate_w + (long long)i * HH * INTERi, gbuf, nullptr,
                BATCH * QQ, INTERi, HH, HH, INTERi, 1, INTERi, 1.0f);
    launch_gemm(stream, x, up_w + (long long)i * HH * INTERi, ubuf, nullptr,
                BATCH * QQ, INTERi, HH, HH, INTERi, 1, INTERi, 1.0f);
    const long long mlpElems = (long long)BATCH * QQ * INTERi;
    silu_mul_kernel<<<ceilDiv(mlpElems, 256), 256, 0, stream>>>(gbuf, ubuf, mlpElems);
    launch_gemm(stream, gbuf, down_w + (long long)i * INTERi * HH, h, h,
                BATCH * QQ, HH, INTERi, INTERi, HH, 1, HH, 1.0f);
  }

  // out = rms_norm(h, final_norm_w)
  rmsnorm_kernel<<<BATCH * QQ, 256, 0, stream>>>(
      h, out, nullptr, fnorm_w, HH, QQ,
      (long long)QQ * HH, HH, (long long)QQ * HH, HH, 0, 0);
  (void)in_sizes; (void)n_in; (void)out_size; (void)ws_size;
}
